// Attention_22110491639971
// MI455X (gfx1250) — compile-verified, matrix operand order checked by reading, not run on hardware
//
#include <hip/hip_runtime.h>
#include <hip/hip_bf16.h>
#include <math.h>

typedef __attribute__((ext_vector_type(16))) __bf16 v16bf;
typedef __attribute__((ext_vector_type(8)))  __bf16 v8bf;
typedef __attribute__((ext_vector_type(8)))  float  v8f;
typedef __attribute__((ext_vector_type(4)))  unsigned int u32x4;
typedef __attribute__((ext_vector_type(8)))  int i32x8;
typedef __attribute__((ext_vector_type(4)))  int i32x4;

#define RADIUS 50
#define HEADS  4
#define Bsz    2
#define Cch    256
#define Lseq   1536
#define Dh     64
#define NBL    (Bsz * Lseq)      /* 3072 */
#define NTOT   (Bsz * Cch * Lseq)
#define NEG_BIG (-1e30f)
#define SSHIFT 56                /* band window start: t0 - 56, 16B aligned */
#define JOFF   (RADIUS + SSHIFT) /* j = JOFF + t_local - s_local = 106 + ... */

#if defined(__has_builtin)
#if __has_builtin(__builtin_amdgcn_tensor_load_to_lds)
#define HAVE_TDM 1
#endif
#endif
#ifndef HAVE_TDM
#define HAVE_TDM 0
#endif

// ---------------------------------------------------------------------------
// WMMA helpers (gfx1250, wave32). A-frag: lane%16 = M row, halfwords 0..7 are
// K = hb..hb+7 and halfwords 8..15 are K = hb+16..hb+23, hb = (lane<16)?0:8.
// B-frag mirrors with lane%16 = N. C/D-frag: lane%16 = N, VGPR v = M (+8 for
// upper half-wave).
// ---------------------------------------------------------------------------
__device__ __forceinline__ v16bf load_frag(const __bf16* rowptr, int hb) {
  v8bf r0 = *reinterpret_cast<const v8bf*>(rowptr + hb);
  v8bf r1 = *reinterpret_cast<const v8bf*>(rowptr + hb + 16);
  v16bf f;
#pragma unroll
  for (int i = 0; i < 8; ++i) { f[i] = r0[i]; f[i + 8] = r1[i]; }
  return f;
}

__device__ __forceinline__ v8f wmma_bf16(v16bf a, v16bf b, v8f c) {
  return __builtin_amdgcn_wmma_f32_16x16x32_bf16(false, a, false, b,
                                                 (short)0, c, false, false);
}

// ---------------------------------------------------------------------------
// Tensor Data Mover: 2D tile load Global -> LDS, bf16 elements.
// D# built per CDNA5 ISA (group0: 4 dwords, group1: 8).
// tensor_x / tensor_rows clip the tile at the tensor edge (OOB reads -> 0).
// ---------------------------------------------------------------------------
__device__ __forceinline__ void tdm_load_2d(unsigned lds_byte_off,
                                            const void* gaddr,
                                            unsigned tile_x, unsigned tile_rows,
                                            unsigned long long tensor_x,
                                            unsigned long long tensor_rows,
                                            unsigned long long stride_elems) {
#if HAVE_TDM
  unsigned long long ga = (unsigned long long)(uintptr_t)gaddr;
  u32x4 g0;
  g0[0] = 1u;                                    // count=1, user descriptor
  g0[1] = lds_byte_off;                          // lds_addr (bytes)
  g0[2] = (unsigned)(ga & 0xffffffffu);          // global_addr[31:0]
  g0[3] = (unsigned)((ga >> 32) & 0x01ffffffu)   // global_addr[56:32]
          | (2u << 30);                          // type = 2 ("image")
  i32x8 g1;
  g1[0] = (int)(1u << 16);                       // data_size = 1 -> 2 bytes
  g1[1] = (int)((tensor_x & 0xffffu) << 16);     // tensor_dim0[15:0]
  g1[2] = (int)(((tensor_x >> 16) & 0xffffu) |
                ((tensor_rows & 0xffffu) << 16));// dim0[31:16] | dim1[15:0]
  g1[3] = (int)(((tensor_rows >> 16) & 0xffffu) |
                ((tile_x & 0xffffu) << 16));     // dim1[31:16] | tile_dim0
  g1[4] = (int)(tile_rows & 0xffffu);            // tile_dim1 (tile_dim2 = 0)
  g1[5] = (int)(stride_elems & 0xffffffffu);     // tensor_dim0_stride[31:0]
  g1[6] = (int)((stride_elems >> 32) & 0xffffu); // stride[47:32]
  g1[7] = 0;
  i32x4 z4 = {0, 0, 0, 0};
#if __clang_major__ >= 23
  i32x8 z8 = {0, 0, 0, 0, 0, 0, 0, 0};
  __builtin_amdgcn_tensor_load_to_lds(g0, g1, z4, z4, z8, 0);
#else
  __builtin_amdgcn_tensor_load_to_lds(g0, g1, z4, z4, 0);
#endif
#endif
}

__device__ __forceinline__ unsigned lds_offset_of(const void* p) {
  return (unsigned)(uintptr_t)p;  // LDS aperture: low 32 bits = byte offset
}

// ---------------------------------------------------------------------------
// Converters
// ---------------------------------------------------------------------------
__global__ void k_f32_to_bf16(const float* __restrict__ src,
                              __bf16* __restrict__ dst, int n) {
  int i = blockIdx.x * blockDim.x + threadIdx.x;
  if (i < n) dst[i] = (__bf16)src[i];
}

// x[b][c][l] -> xT[(b*L + l)*C + c]  (bf16, c-contiguous)
__global__ void k_transpose_x(const float* __restrict__ x,
                              __bf16* __restrict__ xT) {
  int i = blockIdx.x * blockDim.x + threadIdx.x;
  if (i >= NTOT) return;
  int l = i % Lseq;
  int c = (i / Lseq) % Cch;
  int b = i / (Lseq * Cch);
  xT[(size_t)(b * Lseq + l) * Cch + c] = (__bf16)x[i];
}

// emb[j][c] (101x64) -> embN[128][64] (zero-padded) and embT[64][128]
__global__ void k_prep_emb(const float* __restrict__ emb,
                           __bf16* __restrict__ embN,
                           __bf16* __restrict__ embT) {
  int i = blockIdx.x * blockDim.x + threadIdx.x;
  if (i >= 128 * 64) return;
  int c = i % 64;
  int j = i / 64;
  float v = (j <= 2 * RADIUS) ? emb[j * Dh + c] : 0.0f;
  embN[j * 64 + c]  = (__bf16)v;
  embT[c * 128 + j] = (__bf16)v;
}

// ---------------------------------------------------------------------------
// Generic bf16 GEMM: out[o,n] = sum_c W[o,c] * Bm[n,c] + bias[o]
// One wave computes a 16(M) x 64(N) tile; K=256 in 8 WMMA steps (32 wmma).
// MODE 0: bf16 out[n*256 + o]        MODE 1: bf16 out[b*C*L + o*L + l]
// MODE 2: f32  out[b*C*L + o*L + l]
// ---------------------------------------------------------------------------
template <int MODE>
__global__ void __launch_bounds__(32)
k_gemm(const __bf16* __restrict__ W, const __bf16* __restrict__ Bm,
       const float* __restrict__ bias, void* __restrict__ outv) {
  const int lane = threadIdx.x;
  const int nl = lane & 15;
  const int hi = lane >> 4;
  const int hb = hi * 8;
  const int m0 = blockIdx.x * 16;
  const int n0 = blockIdx.y * 64;

  __builtin_prefetch(Bm + (size_t)(n0 + lane) * Cch, 0, 0);

  v8f acc[4] = {};
#pragma unroll
  for (int k0 = 0; k0 < Cch; k0 += 32) {
    v16bf a = load_frag(W + (size_t)(m0 + nl) * Cch + k0, hb);
#pragma unroll
    for (int j = 0; j < 4; ++j) {
      v16bf bfr = load_frag(Bm + (size_t)(n0 + j * 16 + nl) * Cch + k0, hb);
      acc[j] = wmma_bf16(a, bfr, acc[j]);
    }
  }

  const int mbase = m0 + hb;
#pragma unroll
  for (int j = 0; j < 4; ++j) {
    const int n = n0 + j * 16 + nl;
    if (MODE == 0) {
      __bf16* out = (__bf16*)outv;
      v8bf st;
#pragma unroll
      for (int v = 0; v < 8; ++v)
        st[v] = (__bf16)(acc[j][v] + bias[mbase + v]);
      *reinterpret_cast<v8bf*>(out + (size_t)n * Cch + mbase) = st;
    } else if (MODE == 1) {
      __bf16* out = (__bf16*)outv;
      const int bb = n / Lseq, l = n % Lseq;
#pragma unroll
      for (int v = 0; v < 8; ++v)
        out[(size_t)bb * Cch * Lseq + (size_t)(mbase + v) * Lseq + l] =
            (__bf16)(acc[j][v] + bias[mbase + v]);
    } else {
      float* out = (float*)outv;
      const int bb = n / Lseq, l = n % Lseq;
#pragma unroll
      for (int v = 0; v < 8; ++v)
        out[(size_t)bb * Cch * Lseq + (size_t)(mbase + v) * Lseq + l] =
            acc[j][v] + bias[mbase + v];
    }
  }
}

// ---------------------------------------------------------------------------
// Banded attention. One wave per (b, h, 16-row t-tile).
// Band window: s in [t0 - 56, t0 + 72), 128 wide, 16B aligned.
// ---------------------------------------------------------------------------
__global__ void __launch_bounds__(32)
k_attn(const __bf16* __restrict__ qb, const __bf16* __restrict__ kb,
       const __bf16* __restrict__ cont, const __bf16* __restrict__ embN,
       const __bf16* __restrict__ embT, __bf16* __restrict__ attnT) {
  __shared__ __bf16 kls[128 * 64];    // [s_local][c]
  __shared__ __bf16 cls[64 * 128];    // [c][s_local]
  __shared__ float  qels[16 * 128];   // [t][j]
  __shared__ float  dls[16 * 128];    // [t][s_local]
  __shared__ __bf16 wls[16 * 128];    // softmax weights [t][s_local]
  __shared__ __bf16 wbls[16 * 128];   // wbar [t][j]

  const int lane = threadIdx.x;
  const int nl = lane & 15;
  const int hi = lane >> 4;
  const int hb = hi * 8;

  int tile = blockIdx.x;
  const int tt = tile % (Lseq / 16); tile /= (Lseq / 16);
  const int h = tile % HEADS;
  const int b = tile / HEADS;
  const int t0 = tt * 16;
  const int s_start = t0 - SSHIFT;

  const __bf16* qrow0 = qb + (size_t)(b * Lseq + t0 + nl) * Cch + h * Dh;
  __builtin_prefetch(qrow0, 0, 0);

  // ---- stage K band slice [s_local][c] ----
  const __bf16* ksrc = kb + (size_t)(b * Lseq) * Cch + h * Dh;  // row s base
  const int first = (s_start < 0) ? 0 : s_start;
  const int pre = first - s_start;  // top-edge rows to pre-zero
#if HAVE_TDM
  for (int i = lane; i < pre * 64; i += 32) kls[i] = (__bf16)0.0f;
  tdm_load_2d(lds_offset_of(kls) + (unsigned)pre * 64u * 2u,
              ksrc + (size_t)first * Cch,
              /*tile_x=*/64u, /*tile_rows=*/(unsigned)(128 - pre),
              /*tensor_x=*/64ull,
              /*tensor_rows=*/(unsigned long long)(Lseq - first),
              /*stride=*/(unsigned long long)Cch);
#else
  for (int r = lane; r < 128; r += 32) {
    const int s = s_start + r;
    if (s >= 0 && s < Lseq) {
      const __bf16* src = ksrc + (size_t)s * Cch;
#pragma unroll
      for (int q8 = 0; q8 < 8; ++q8)
        *reinterpret_cast<v8bf*>(&kls[r * 64 + q8 * 8]) =
            *reinterpret_cast<const v8bf*>(src + q8 * 8);
    } else {
      for (int c = 0; c < 64; ++c) kls[r * 64 + c] = (__bf16)0.0f;
    }
  }
#endif

  // ---- stage content band slice [c][s_local] ----
  const __bf16* csrc = cont + (size_t)b * Cch * Lseq + (size_t)(h * Dh) * Lseq;
  bool cls_manual = true;
#if HAVE_TDM
  if (s_start >= 0) {
    tdm_load_2d(lds_offset_of(cls), csrc + s_start,
                /*tile_x=*/128u, /*tile_rows=*/64u,
                /*tensor_x=*/(unsigned long long)(Lseq - s_start),
                /*tensor_rows=*/64ull,
                /*stride=*/(unsigned long long)Lseq);
    cls_manual = false;
  }
#endif
  if (cls_manual) {
    for (int c = lane; c < 64; c += 32) {
      const __bf16* src = csrc + (size_t)c * Lseq;
      __bf16* dst = &cls[c * 128];
#pragma unroll
      for (int q = 0; q < 16; ++q) {
        const int s0 = s_start + q * 8;
        if (s0 >= 0 && s0 + 8 <= Lseq) {
          *reinterpret_cast<v8bf*>(dst + q * 8) =
              *reinterpret_cast<const v8bf*>(src + s0);
        } else {
          for (int e = 0; e < 8; ++e) {
            const int s = s0 + e;
            dst[q * 8 + e] = (s >= 0 && s < Lseq) ? src[s] : (__bf16)0.0f;
          }
        }
      }
    }
  }
#if HAVE_TDM
  __builtin_amdgcn_s_wait_tensorcnt(0);
#endif
  __syncthreads();

  // ---- Q fragments (A-matrix, M=t, K=c in two 32-chunks) ----
  const v16bf aq0 = load_frag(qrow0, hb);
  const v16bf aq1 = load_frag(qrow0 + 32, hb);

  // ---- QE[t,j] = Q . emb^T (16 wmma) ----
#pragma unroll
  for (int jn = 0; jn < 8; ++jn) {
    v8f acc = {};
    const __bf16* erow = embN + (size_t)(jn * 16 + nl) * 64;
    acc = wmma_bf16(aq0, load_frag(erow, hb), acc);
    acc = wmma_bf16(aq1, load_frag(erow + 32, hb), acc);
    const int j = jn * 16 + nl;
#pragma unroll
    for (int v = 0; v < 8; ++v) qels[(hb + v) * 128 + j] = acc[v];
  }
  __syncthreads();

  // ---- dots[t,s] = Q.K + 0.3*QE[t, j], j = JOFF + t - s_local ----
#pragma unroll
  for (int sn = 0; sn < 8; ++sn) {
    v8f acc = {};
    const __bf16* krow = &kls[(sn * 16 + nl) * 64];
    acc = wmma_bf16(aq0, load_frag(krow, hb), acc);
    acc = wmma_bf16(aq1, load_frag(krow + 32, hb), acc);
    const int sl = sn * 16 + nl;
    const int s = s_start + sl;
#pragma unroll
    for (int v = 0; v < 8; ++v) {
      const int t = hb + v;
      const int j = JOFF + t - sl;
      const bool band = (j >= 0 && j <= 2 * RADIUS);
      float val = acc[v] + (band ? 0.3f * qels[t * 128 + j] : 0.0f);
      const bool valid = band && (s >= 0 && s < Lseq);
      dls[t * 128 + sl] = valid ? val : NEG_BIG;
    }
  }
  __syncthreads();

  // ---- softmax: lanes (t, t+16) split each 128-wide row ----
  {
    const int t = nl;
    const int sbeg = hi * 64;
    float m = NEG_BIG;
    for (int sl = sbeg; sl < sbeg + 64; ++sl) m = fmaxf(m, dls[t * 128 + sl]);
    m = fmaxf(m, __shfl_xor(m, 16, 32));
    float ssum = 0.0f;
    for (int sl = sbeg; sl < sbeg + 64; ++sl)
      ssum += __expf(dls[t * 128 + sl] - m);
    ssum += __shfl_xor(ssum, 16, 32);
    const float inv = 1.0f / ssum;
    for (int sl = sbeg; sl < sbeg + 64; ++sl)
      wls[t * 128 + sl] = (__bf16)(__expf(dls[t * 128 + sl] - m) * inv);
    // wbar[t][j] = w[t, sl = JOFF + t - j] for j in [0, 2R], else 0
    for (int j = sbeg; j < sbeg + 64; ++j) {
      const int sl = JOFF + t - j;
      float w = (j <= 2 * RADIUS && sl >= 0 && sl < 128)
                    ? __expf(dls[t * 128 + sl] - m) * inv
                    : 0.0f;
      wbls[t * 128 + j] = (__bf16)w;
    }
  }
  __syncthreads();

  // ---- out[c,t] = content.W^T + 0.3 * emb^T.Wbar^T (32 wmma) ----
#pragma unroll
  for (int cm = 0; cm < 4; ++cm) {
    v8f acc1 = {};
    v8f acc2 = {};
#pragma unroll
    for (int kk = 0; kk < 4; ++kk) {
      const v16bf aC = load_frag(&cls[(cm * 16 + nl) * 128 + kk * 32], hb);
      const v16bf bW = load_frag(&wls[nl * 128 + kk * 32], hb);
      acc1 = wmma_bf16(aC, bW, acc1);
      const v16bf aE = load_frag(embT + (size_t)(cm * 16 + nl) * 128 + kk * 32, hb);
      const v16bf bW2 = load_frag(&wbls[nl * 128 + kk * 32], hb);
      acc2 = wmma_bf16(aE, bW2, acc2);
    }
    v8bf st;
#pragma unroll
    for (int v = 0; v < 8; ++v)
      st[v] = (__bf16)(acc1[v] + 0.3f * acc2[v]);
    *reinterpret_cast<v8bf*>(attnT + (size_t)(b * Lseq + t0 + nl) * Cch +
                             h * Dh + cm * 16 + hb) = st;
  }
}

// ---------------------------------------------------------------------------
// BatchNorm statistics: one block per channel, reduce over (B, L)
// ---------------------------------------------------------------------------
__global__ void k_bnstats(const float* __restrict__ fout,
                          float* __restrict__ mean, float* __restrict__ rstd) {
  __shared__ float s1[256];
  __shared__ float s2[256];
  const int o = blockIdx.x;
  float a = 0.0f, q = 0.0f;
  for (int i = threadIdx.x; i < NBL; i += blockDim.x) {
    const int bb = i / Lseq, l = i % Lseq;
    const float v = fout[(size_t)bb * Cch * Lseq + (size_t)o * Lseq + l];
    a += v;
    q += v * v;
  }
  s1[threadIdx.x] = a;
  s2[threadIdx.x] = q;
  __syncthreads();
  for (int st = 128; st > 0; st >>= 1) {
    if ((int)threadIdx.x < st) {
      s1[threadIdx.x] += s1[threadIdx.x + st];
      s2[threadIdx.x] += s2[threadIdx.x + st];
    }
    __syncthreads();
  }
  if (threadIdx.x == 0) {
    const float mu = s1[0] / (float)NBL;
    const float var = s2[0] / (float)NBL - mu * mu;
    mean[o] = mu;
    rstd[o] = rsqrtf(var + 1e-5f);
  }
}

__global__ void k_final(const float* __restrict__ fout,
                        const float* __restrict__ mean,
                        const float* __restrict__ rstd,
                        const float* __restrict__ gamma,
                        const float* __restrict__ beta,
                        const float* __restrict__ scale,
                        float* __restrict__ out) {
  int i = blockIdx.x * blockDim.x + threadIdx.x;
  if (i >= NTOT) return;
  const int o = (i / Lseq) % Cch;
  float v = (fout[i] - mean[o]) * rstd[o] * gamma[o] + beta[o];
  out[i] = fmaxf(v, 0.0f) * scale[o];
}

// ---------------------------------------------------------------------------
extern "C" void kernel_launch(void* const* d_in, const int* in_sizes, int n_in,
                              void* d_out, int out_size, void* d_ws,
                              size_t ws_size, hipStream_t stream) {
  (void)in_sizes; (void)n_in; (void)out_size; (void)ws_size;

  const float* x     = (const float*)d_in[0];
  const float* Wc    = (const float*)d_in[1];
  const float* bc    = (const float*)d_in[2];
  const float* Wq    = (const float*)d_in[3];
  const float* bq    = (const float*)d_in[4];
  const float* Wk    = (const float*)d_in[5];
  const float* bk    = (const float*)d_in[6];
  const float* emb   = (const float*)d_in[7];
  const float* Wf    = (const float*)d_in[8];
  const float* bf_   = (const float*)d_in[9];
  const float* gamma = (const float*)d_in[10];
  const float* beta  = (const float*)d_in[11];
  const float* scale = (const float*)d_in[12];

  char* p = (char*)d_ws;
  auto carve = [&p](size_t bytes) -> void* {
    void* r = (void*)p;
    p += (bytes + 255) & ~(size_t)255;
    return r;
  };

  __bf16* xT    = (__bf16*)carve((size_t)NBL * Cch * 2);
  __bf16* Wcb   = (__bf16*)carve((size_t)Cch * Cch * 2);
  __bf16* Wqb   = (__bf16*)carve((size_t)Cch * Cch * 2);
  __bf16* Wkb   = (__bf16*)carve((size_t)Cch * Cch * 2);
  __bf16* Wfb   = (__bf16*)carve((size_t)Cch * Cch * 2);
  __bf16* embN  = (__bf16*)carve((size_t)128 * 64 * 2);
  __bf16* embT  = (__bf16*)carve((size_t)64 * 128 * 2);
  __bf16* qbuf  = (__bf16*)carve((size_t)NBL * Cch * 2);
  __bf16* kbuf  = (__bf16*)carve((size_t)NBL * Cch * 2);
  __bf16* contb = (__bf16*)carve((size_t)NTOT * 2);
  __bf16* attnT = (__bf16*)carve((size_t)NBL * Cch * 2);
  float*  fout  = (float*)carve((size_t)NTOT * 4);
  float*  meanb = (float*)carve((size_t)Cch * 4);
  float*  rstdb = (float*)carve((size_t)Cch * 4);

  const int WW = Cch * Cch;  // 65536
  k_transpose_x<<<(NTOT + 255) / 256, 256, 0, stream>>>(x, xT);
  k_f32_to_bf16<<<(WW + 255) / 256, 256, 0, stream>>>(Wc, Wcb, WW);
  k_f32_to_bf16<<<(WW + 255) / 256, 256, 0, stream>>>(Wq, Wqb, WW);
  k_f32_to_bf16<<<(WW + 255) / 256, 256, 0, stream>>>(Wk, Wkb, WW);
  k_f32_to_bf16<<<(WW + 255) / 256, 256, 0, stream>>>(Wf, Wfb, WW);
  k_prep_emb<<<(128 * 64 + 255) / 256, 256, 0, stream>>>(emb, embN, embT);

  dim3 gg(Cch / 16, NBL / 64);  // 16 x 48
  k_gemm<0><<<gg, 32, 0, stream>>>(Wqb, xT, bq, (void*)qbuf);
  k_gemm<0><<<gg, 32, 0, stream>>>(Wkb, xT, bk, (void*)kbuf);
  k_gemm<1><<<gg, 32, 0, stream>>>(Wcb, xT, bc, (void*)contb);

  k_attn<<<Bsz * HEADS * (Lseq / 16), 32, 0, stream>>>(qbuf, kbuf, contb,
                                                       embN, embT, attnT);

  k_gemm<2><<<gg, 32, 0, stream>>>(Wfb, attnT, bf_, (void*)fout);

  k_bnstats<<<Cch, 256, 0, stream>>>(fout, meanb, rstdb);
  k_final<<<(NTOT + 255) / 256, 256, 0, stream>>>(fout, meanb, rstdb, gamma,
                                                  beta, scale, (float*)d_out);
}
